// GlobalFluxTracker_6932077216156
// MI455X (gfx1250) — matrix operand order repaired and checked by reading, not compiled, not run on hardware
//
#include <hip/hip_runtime.h>
#include <hip/hip_bf16.h>
#include <math.h>

// ---------------------------------------------------------------------------
// GlobalFluxTracker for MI455X (gfx1250, wave32, WMMA)
// Pipeline: pack -> GEMM1(W_mix) -> chunked complex scan -> GEMM2(W_out)
//           -> GEMM3(W_gc) -> GEMM4(W_gate, fused sigmoid) ; bf16 WMMA GEMMs.
// GEMM: block tile 128x256, 8 waves (2x4), wave tile 64x64 = 4x4 WMMA tiles.
// ---------------------------------------------------------------------------

typedef __bf16 bf16_t;
typedef __bf16 v16bf __attribute__((ext_vector_type(16)));
typedef float  v8f   __attribute__((ext_vector_type(8)));

#define BB 4
#define TT 2048
#define DD 1024
#define MM (BB * TT)          // 8192 rows
#define NCHUNK 32
#define CLEN (TT / NCHUNK)    // 64

// ------------------------------ pack kernels -------------------------------

__global__ __launch_bounds__(256) void pack_xcat(
    const float* __restrict__ x_re, const float* __restrict__ x_im,
    bf16_t* __restrict__ Xcat)
{
    size_t idx = (size_t)blockIdx.x * 256 + threadIdx.x;   // < MM * 2*DD
    if (idx >= (size_t)MM * 2 * DD) return;
    size_t m = idx >> 11;            // / (2*DD)
    int    c = (int)(idx & 2047);
    float v = (c < DD) ? x_re[m * DD + c] : x_im[m * DD + (c - DD)];
    Xcat[idx] = (bf16_t)v;
}

// W [K,N] fp32 row-major  ->  WT [N,K] bf16 row-major
__global__ __launch_bounds__(256) void pack_wT(
    const float* __restrict__ W, bf16_t* __restrict__ WT, int K, int N)
{
    size_t idx = (size_t)blockIdx.x * 256 + threadIdx.x;
    if (idx >= (size_t)K * N) return;
    int k = (int)(idx / N);
    int n = (int)(idx % N);
    WT[(size_t)n * K + k] = (bf16_t)W[idx];
}

__global__ __launch_bounds__(256) void cvt_f32_to_bf16(
    const float* __restrict__ src, bf16_t* __restrict__ dst, size_t total)
{
    size_t idx = (size_t)blockIdx.x * 256 + threadIdx.x;
    if (idx < total) dst[idx] = (bf16_t)src[idx];
}

// copy src (fp32, rows of d_out stride 3*DD, cols [0,2*DD)) into bf16
// gate_in cols [2*DD, 4*DD)
__global__ __launch_bounds__(256) void cvt_src_to_gatein(
    const float* __restrict__ out, bf16_t* __restrict__ gate_in)
{
    size_t idx = (size_t)blockIdx.x * 256 + threadIdx.x;   // < MM * 2*DD
    if (idx >= (size_t)MM * 2 * DD) return;
    size_t m = idx >> 11;
    int    c = (int)(idx & 2047);
    gate_in[m * (4 * DD) + 2 * DD + c] = (bf16_t)out[m * (3 * DD) + c];
}

// ------------------------- bf16 WMMA tiled GEMM ----------------------------
// C[M,N] (fp32) = A[M,K](bf16, row stride lda) * Bt[N,K]^T (bf16) + bias[N]
// Block tile 128x256, K-step 32, 8 waves (2 M x 4 N), 4x4 WMMA tiles / wave.
__global__ __launch_bounds__(256) void gemm_bf16_wmma(
    const bf16_t* __restrict__ A, int lda,
    const bf16_t* __restrict__ Bt,
    const float* __restrict__ bias,
    float* __restrict__ C, int ldc,
    int K, int act)
{
    __shared__ bf16_t lA[128 * 32];   //  8 KB
    __shared__ bf16_t lB[256 * 32];   // 16 KB

    const int tid   = threadIdx.x;
    const int lane  = tid & 31;
    const int wave  = tid >> 5;          // 0..7
    const int wm    = wave >> 2;         // 0..1  (M dir, 64 rows each)
    const int wn    = wave & 3;          // 0..3  (N dir, 64 cols each)
    const int m16   = lane & 15;
    const int khalf = lane >> 4;         // 0 or 1
    const size_t bm = (size_t)blockIdx.x * 128;
    const size_t bn = (size_t)blockIdx.y * 256;

    v8f acc[4][4] = {};

    const int arow  = tid >> 1;          // 0..127  (A coop load row)
    const int ahalf = tid & 1;           // 0..1 : 16 bf16 (32B) each

    for (int k0 = 0; k0 < K; k0 += 32) {
        // cooperative tile load: A 128x32, B 256x32 (bf16)
        {
            const uint4* ga = (const uint4*)(A + (bm + arow) * (size_t)lda + k0 + ahalf * 16);
            uint4* da = (uint4*)(lA + arow * 32 + ahalf * 16);
            da[0] = ga[0];
            da[1] = ga[1];
            const uint4* gb = (const uint4*)(Bt + (bn + tid) * (size_t)K + k0);
            uint4* db = (uint4*)(lB + tid * 32);
            db[0] = gb[0];
            db[1] = gb[1];
            db[2] = gb[2];
            db[3] = gb[3];
        }
        __syncthreads();

        // A fragment (16x32, ISA layout): lane holds row m16, runs
        // K=[khalf*8, +8) and K=[16+khalf*8, +8)
        v16bf af[4];
        #pragma unroll
        for (int i = 0; i < 4; ++i) {
            const bf16_t* p = lA + (wm * 64 + i * 16 + m16) * 32 + khalf * 8;
            uint4* fp = (uint4*)&af[i];
            fp[0] = *(const uint4*)p;
            fp[1] = *(const uint4*)(p + 16);
        }
        // B fragment (32x16): lane holds column m16, K=[khalf*8, +8) and K=[16+khalf*8, +8)
        v16bf bfg[4];
        #pragma unroll
        for (int j = 0; j < 4; ++j) {
            const bf16_t* p = lB + (wn * 64 + j * 16 + m16) * 32 + khalf * 8;
            uint4* fp = (uint4*)&bfg[j];
            fp[0] = *(const uint4*)p;
            fp[1] = *(const uint4*)(p + 16);
        }

        #pragma unroll
        for (int i = 0; i < 4; ++i)
            #pragma unroll
            for (int j = 0; j < 4; ++j)
                acc[i][j] = __builtin_amdgcn_wmma_f32_16x16x32_bf16(
                    false, af[i], false, bfg[j], (short)0, acc[i][j], false, false);
        __syncthreads();
    }

    // epilogue: C/D 16x16 layout — VGPR r: lanes0-15 M=r, lanes16-31 M=8+r
    #pragma unroll
    for (int i = 0; i < 4; ++i) {
        #pragma unroll
        for (int j = 0; j < 4; ++j) {
            size_t row0 = bm + wm * 64 + i * 16 + khalf * 8;
            size_t col  = bn + wn * 64 + j * 16 + m16;
            float bv = bias ? bias[col] : 0.0f;
            #pragma unroll
            for (int r = 0; r < 8; ++r) {
                float v = acc[i][j][r] + bv;
                if (act == 1) v = 0.98f / (1.0f + __expf(-v)) + 0.01f;
                C[(row0 + r) * (size_t)ldc + col] = v;
            }
        }
    }
}

// -------------------------- chunked complex scan ---------------------------

__device__ __forceinline__ void step_decay_forcing(
    float lre, float lim, float dtr,
    float& are, float& aim, float& fre, float& fim)
{
    float xr = lre * dtr, xi = lim * dtr;       // exp_arg = lam * dt_ratio
    float s, c;
    __sincosf(xi, &s, &c);
    float er = __expf(xr);
    are = er * c;                               // decay = exp(arg)
    aim = er * s;
    float mag2 = xr * xr + xi * xi;
    float pre, pim;
    if (mag2 > 1e-14f) {                        // |arg| > 1e-7
        // cexpm1(z): re = expm1(x)cos(y) - 2 sin^2(y/2) ; im = e^x sin(y)
        float em = expm1f(xr);
        float sh = __sinf(xi * 0.5f);
        float cr = em * c - 2.0f * sh * sh;
        float ci = er * s;
        float inv = 1.0f / mag2;                // divide by z
        pre = (cr * xr + ci * xi) * inv;
        pim = (ci * xr - cr * xi) * inv;
    } else {                                    // 1 + z/2 + z^2/6
        pre = 1.0f + 0.5f * xr + (xr * xr - xi * xi) * (1.0f / 6.0f);
        pim = 0.5f * xi + (2.0f * xr * xi) * (1.0f / 6.0f);
    }
    fre = dtr * pre;                            // forcing = dt_ratio * phi1
    fim = dtr * pim;
}

// pass 1: per (b, chunk, d) compute chunk summary (A, B) with h_in = 0
__global__ __launch_bounds__(256) void scan_pass1(
    const float* __restrict__ xin, const float* __restrict__ dt_seq,
    const float* __restrict__ dre, const float* __restrict__ dim_,
    float4* __restrict__ chunkAB)
{
    int d  = blockIdx.x * 256 + threadIdx.x;
    int ch = blockIdx.y, b = blockIdx.z;
    float lre = -log1pf(__expf(dre[d]));        // -softplus(decay_re)
    float lim = dim_[d];
    float Are = 1.0f, Aim = 0.0f, Bre = 0.0f, Bim = 0.0f;
    int t0 = ch * CLEN;
    for (int i = 0; i < CLEN; ++i) {
        int t = t0 + i;
        float dtr = dt_seq[b * TT + t];
        float are, aim, fre, fim;
        step_decay_forcing(lre, lim, dtr, are, aim, fre, fim);
        const float* row = xin + (size_t)(b * TT + t) * (2 * DD);
        float xr = row[d], xi = row[DD + d];
        float sr = xr * fre - xi * fim;
        float si = xr * fim + xi * fre;
        float nAre = are * Are - aim * Aim;
        float nAim = are * Aim + aim * Are;
        float nBre = are * Bre - aim * Bim + sr;
        float nBim = are * Bim + aim * Bre + si;
        Are = nAre; Aim = nAim; Bre = nBre; Bim = nBim;
    }
    chunkAB[(size_t)(b * NCHUNK + ch) * DD + d] = make_float4(Are, Aim, Bre, Bim);
}

// pass 2: per (b, d) chain the 32 chunk summaries -> h at each chunk start
__global__ __launch_bounds__(256) void scan_pass2(
    const float4* __restrict__ chunkAB,
    const float* __restrict__ h0re, const float* __restrict__ h0im,
    float2* __restrict__ carry)
{
    int d = blockIdx.x * 256 + threadIdx.x;
    int b = blockIdx.y;
    float hr = h0re[d], hi = h0im[d];
    for (int ch = 0; ch < NCHUNK; ++ch) {
        carry[(size_t)(b * NCHUNK + ch) * DD + d] = make_float2(hr, hi);
        float4 ab = chunkAB[(size_t)(b * NCHUNK + ch) * DD + d];
        float nr = ab.x * hr - ab.y * hi + ab.z;
        float ni = ab.x * hi + ab.y * hr + ab.w;
        hr = nr; hi = ni;
    }
}

// pass 3: per (b, chunk, d) rerun chunk from carried h, emit bf16 state
__global__ __launch_bounds__(256) void scan_pass3(
    const float* __restrict__ xin, const float* __restrict__ dt_seq,
    const float* __restrict__ dre, const float* __restrict__ dim_,
    const float2* __restrict__ carry, bf16_t* __restrict__ gate_in)
{
    int d  = blockIdx.x * 256 + threadIdx.x;
    int ch = blockIdx.y, b = blockIdx.z;
    float lre = -log1pf(__expf(dre[d]));
    float lim = dim_[d];
    float2 h0 = carry[(size_t)(b * NCHUNK + ch) * DD + d];
    float hr = h0.x, hi = h0.y;
    int t0 = ch * CLEN;
    for (int i = 0; i < CLEN; ++i) {
        int t = t0 + i;
        float dtr = dt_seq[b * TT + t];
        float are, aim, fre, fim;
        step_decay_forcing(lre, lim, dtr, are, aim, fre, fim);
        const float* row = xin + (size_t)(b * TT + t) * (2 * DD);
        float xr = row[d], xi = row[DD + d];
        float sr = xr * fre - xi * fim;
        float si = xr * fim + xi * fre;
        float nr = are * hr - aim * hi + sr;
        float ni = are * hi + aim * hr + si;
        hr = nr; hi = ni;
        bf16_t* grow = gate_in + (size_t)(b * TT + t) * (4 * DD);
        grow[d]      = (bf16_t)hr;
        grow[DD + d] = (bf16_t)hi;
    }
}

// ------------------------------- launcher ----------------------------------

extern "C" void kernel_launch(void* const* d_in, const int* in_sizes, int n_in,
                              void* d_out, int out_size, void* d_ws, size_t ws_size,
                              hipStream_t stream)
{
    const float* x_re     = (const float*)d_in[0];
    const float* x_im     = (const float*)d_in[1];
    const float* dt_seq   = (const float*)d_in[2];
    const float* decay_re = (const float*)d_in[3];
    const float* decay_im = (const float*)d_in[4];
    const float* h0_re    = (const float*)d_in[5];
    const float* h0_im    = (const float*)d_in[6];
    const float* W_mix    = (const float*)d_in[7];
    const float* b_mix    = (const float*)d_in[8];
    const float* W_out    = (const float*)d_in[9];
    const float* b_out    = (const float*)d_in[10];
    const float* W_gc     = (const float*)d_in[11];
    const float* b_gc     = (const float*)d_in[12];
    const float* W_gate   = (const float*)d_in[13];
    const float* b_gate   = (const float*)d_in[14];
    float* out = (float*)d_out;
    (void)in_sizes; (void)n_in; (void)out_size; (void)ws_size;

    // workspace carve-out (~209 MB total, with reuse)
    char* ws = (char*)d_ws;
    size_t off = 0;
    auto carve = [&](size_t bytes) -> char* {
        char* p = ws + off;
        off = (off + bytes + 255) & ~(size_t)255;
        return p;
    };
    bf16_t* Xcat    = (bf16_t*)carve((size_t)MM * 2 * DD * 2); // later: gate_ctx bf16
    bf16_t* WmixT   = (bf16_t*)carve((size_t)2 * DD * 2 * DD * 2);
    bf16_t* WoutT   = (bf16_t*)carve((size_t)2 * DD * 2 * DD * 2);
    bf16_t* WgcT    = (bf16_t*)carve((size_t)4 * DD * 2 * DD * 2);
    bf16_t* WgateT  = (bf16_t*)carve((size_t)2 * DD * DD * 2);
    float*  xin     = (float*)carve((size_t)MM * 2 * DD * 4);  // later: gate_ctx f32
    bf16_t* gate_in = (bf16_t*)carve((size_t)MM * 4 * DD * 2);
    float4* chunkAB = (float4*)carve((size_t)BB * NCHUNK * DD * sizeof(float4));
    float2* carry   = (float2*)carve((size_t)BB * NCHUNK * DD * sizeof(float2));
    float*  gctx_f  = xin;          // reuse after scan consumed xin
    bf16_t* gctx_b  = Xcat;         // reuse after GEMM1 consumed Xcat

    const size_t n2d = (size_t)MM * 2 * DD;     // 16,777,216

    // 1) pack inputs / transpose+convert weights
    pack_xcat<<<dim3((unsigned)(n2d / 256)), dim3(256), 0, stream>>>(x_re, x_im, Xcat);
    pack_wT<<<dim3((unsigned)((size_t)2*DD*2*DD / 256)), dim3(256), 0, stream>>>(W_mix,  WmixT,  2*DD, 2*DD);
    pack_wT<<<dim3((unsigned)((size_t)2*DD*2*DD / 256)), dim3(256), 0, stream>>>(W_out,  WoutT,  2*DD, 2*DD);
    pack_wT<<<dim3((unsigned)((size_t)4*DD*2*DD / 256)), dim3(256), 0, stream>>>(W_gc,   WgcT,   4*DD, 2*DD);
    pack_wT<<<dim3((unsigned)((size_t)2*DD*DD   / 256)), dim3(256), 0, stream>>>(W_gate, WgateT, 2*DD, DD);

    // 2) GEMM1: x_in = Xcat @ W_mix + b_mix        [8192 x 2048], K=2048
    gemm_bf16_wmma<<<dim3(MM/128, (2*DD)/256), dim3(256), 0, stream>>>(
        Xcat, 2*DD, WmixT, b_mix, xin, 2*DD, 2*DD, 0);

    // 3) chunked complex scan -> state (bf16) into gate_in cols [0, 2D)
    scan_pass1<<<dim3(DD/256, NCHUNK, BB), dim3(256), 0, stream>>>(
        xin, dt_seq, decay_re, decay_im, chunkAB);
    scan_pass2<<<dim3(DD/256, BB), dim3(256), 0, stream>>>(
        chunkAB, h0_re, h0_im, carry);
    scan_pass3<<<dim3(DD/256, NCHUNK, BB), dim3(256), 0, stream>>>(
        xin, dt_seq, decay_re, decay_im, carry, gate_in);

    // 4) GEMM2: src = state @ W_out + b_out  -> d_out cols [0, 2D), ldc=3D
    gemm_bf16_wmma<<<dim3(MM/128, (2*DD)/256), dim3(256), 0, stream>>>(
        gate_in, 4*DD, WoutT, b_out, out, 3*DD, 2*DD, 0);

    // 5) src (fp32 in d_out) -> bf16 into gate_in cols [2D, 4D)
    cvt_src_to_gatein<<<dim3((unsigned)(n2d / 256)), dim3(256), 0, stream>>>(out, gate_in);

    // 6) GEMM3: gate_ctx = [state|src] @ W_gc + b_gc   K=4096
    gemm_bf16_wmma<<<dim3(MM/128, (2*DD)/256), dim3(256), 0, stream>>>(
        gate_in, 4*DD, WgcT, b_gc, gctx_f, 2*DD, 4*DD, 0);

    // 7) gate_ctx -> bf16
    cvt_f32_to_bf16<<<dim3((unsigned)(n2d / 256)), dim3(256), 0, stream>>>(gctx_f, gctx_b, n2d);

    // 8) GEMM4: gate = sigmoid(gate_ctx @ W_gate + b_gate)*0.98+0.01
    //    -> d_out cols [2D, 3D), ldc=3D
    gemm_bf16_wmma<<<dim3(MM/128, DD/256), dim3(256), 0, stream>>>(
        gctx_b, 2*DD, WgateT, b_gate, out + 2*DD, 3*DD, 2*DD, 1);
}
